// MemoryModule_6425271075538
// MI455X (gfx1250) — matrix operand order repaired and checked by reading, not compiled, not run on hardware
//
#include <hip/hip_runtime.h>
#include <math.h>

#define HDIM 1024
#define MSLOT 4096
#define WD 64
#define RH 4
#define NWH 2
#define NRM 5
#define EPSX 0.001f

// ---------------- workspace float offsets ----------------
enum {
  WS_WV = 0,            // 128
  WS_EV = 128,          // 128 (sigmoid)
  WS_FG = 256,          // 4   (sigmoid)
  WS_AG = 260,          // 2   (sigmoid)
  WS_WG = 262,          // 2   (sigmoid)
  WS_RM = 264,          // 20  (sigmoid)
  WS_WK = 284,          // 128 (sigmoid)
  WS_WSTR = 412,        // 2
  WS_RK = 416,          // 256
  WS_WLOGIT = 704,                    // NWH*MSLOT
  WS_WCONT  = WS_WLOGIT + NWH*MSLOT,  // NWH*MSLOT
  WS_U      = WS_WCONT  + NWH*MSLOT,  // MSLOT
  WS_ALLOC  = WS_U      + MSLOT,      // NWH*MSLOT
  WS_RLOGIT = WS_ALLOC  + NWH*MSLOT,  // RH*MSLOT
  WS_MN0    = WS_RLOGIT + RH*MSLOT,   // MSLOT (orig memory row norms)
  WS_MN1    = WS_MN0    + MSLOT,      // MSLOT (mem_new row norms)
  WS_WKP    = WS_MN1    + MSLOT,      // 16*64 zero-padded write keys
  WS_RKP    = WS_WKP    + 16*WD,      // 16*64 zero-padded read keys
  WS_KNW    = WS_RKP    + 16*WD,      // 8 write key norms
  WS_KNR    = WS_KNW    + 8,          // 8 read key norms
};

// ---------------- output float offsets ----------------
enum {
  O_READW = 0,                         // RH*WD
  O_MEM   = 256,                       // MSLOT*WD
  O_RW    = O_MEM  + MSLOT*WD,         // RH*MSLOT
  O_WW    = O_RW   + RH*MSLOT,         // NWH*MSLOT
  O_LINK  = O_WW   + NWH*MSLOT,        // NWH*MSLOT*MSLOT
  O_PREC  = O_LINK + NWH*MSLOT*MSLOT,  // NWH*MSLOT
  O_U     = O_PREC + NWH*MSLOT,        // MSLOT
};

typedef __attribute__((ext_vector_type(16))) _Float16 v16h;
typedef __attribute__((ext_vector_type(8)))  float    v8f;
typedef __attribute__((ext_vector_type(4)))  float    v4f;   // native vector for NT builtins

// ---- GEMV projections: out[c] = act(x . W[:,c] + b[c]) ----
__global__ void proj_kernel(const float* __restrict__ x, const float* __restrict__ W,
                            const float* __restrict__ b, float* __restrict__ out,
                            int od, int act) {
  __shared__ float red[256];
  int c = blockIdx.x, t = threadIdx.x;
  float s = 0.f;
  for (int h = t; h < HDIM; h += 256) s += x[h] * W[(size_t)h * od + c];
  red[t] = s; __syncthreads();
  for (int o = 128; o > 0; o >>= 1) { if (t < o) red[t] += red[t + o]; __syncthreads(); }
  if (t == 0) {
    float v = red[0] + b[c];
    if (act) v = 1.f / (1.f + __expf(-v));
    out[c] = v;
  }
}

// ---- zero-pad keys to dense 16x64 tiles + precompute key norms ----
__global__ __launch_bounds__(1024)
void padkeys_kernel(float* __restrict__ ws) {
  int t = threadIdx.x;
  int row = t >> 6, k = t & 63;
  float wv = (row < NWH) ? ws[WS_WK + row * WD + k] : 0.f;
  float rv = (row < RH)  ? ws[WS_RK + row * WD + k] : 0.f;
  ws[WS_WKP + t] = wv;
  ws[WS_RKP + t] = rv;
  __syncthreads();
  if (t < 16) {
    int isr = t >> 3;                 // 0 = write keys, 1 = read keys
    int w   = t & 7;
    const float* base = ws + (isr ? WS_RKP : WS_WKP) + w * WD;
    float s = 0.f;
    for (int kk = 0; kk < WD; ++kk) { float v = base[kk]; s += v * v; }
    float n = sqrtf(s);
    if (n == 0.f) n = 1.f;            // padded rows: harmless non-zero
    ws[(isr ? WS_KNR : WS_KNW) + w] = n;
  }
}

// ---- usage update: u = (usage + (1-usage)*(1-prod ww)) * (1 - prod_r fg_r*rw_r) ----
__global__ void usage_kernel(const float* __restrict__ usage, const float* __restrict__ ww_old,
                             const float* __restrict__ rw_old, const float* __restrict__ ws,
                             float* __restrict__ u_out) {
  int m = blockIdx.x * blockDim.x + threadIdx.x;
  if (m >= MSLOT) return;
  float wr = 1.f - ww_old[m] * ww_old[MSLOT + m];
  float u = usage[m] + (1.f - usage[m]) * wr;
  float p = 1.f;
  for (int r = 0; r < RH; ++r) p *= ws[WS_FG + r] * rw_old[r * MSLOT + m];
  u_out[m] = u * (1.f - p);
}

// ---- per-row L2 norms ----
__global__ void rownorm_kernel(const float* __restrict__ mat, float* __restrict__ norms) {
  __shared__ float red[WD];
  int m = blockIdx.x, d = threadIdx.x;
  float v = mat[(size_t)m * WD + d];
  red[d] = v * v; __syncthreads();
  for (int o = 32; o > 0; o >>= 1) { if (d < o) red[d] += red[d + o]; __syncthreads(); }
  if (d == 0) norms[m] = sqrtf(red[0]);
}

// ---- cosine similarity via WMMA f32_16x16x32_f16: keys(16x64 padded) @ mem^T ----
// One wave per 16-column tile. Branchless float4 loads for both A and B.
__global__ void cosine_wmma_kernel(const float* __restrict__ keysp,   // 16x64, zero padded
                                   const float* __restrict__ knorm,   // >=8 norms
                                   int nkeys,
                                   const float* __restrict__ mem,
                                   const float* __restrict__ mnorm,
                                   const float* __restrict__ strength,
                                   float* __restrict__ out) {
  int tile = blockIdx.x;
  int lane = threadIdx.x;                // 0..31
  int mrow = lane & 15;                  // A: M row
  int khl  = (lane >= 16) ? 8 : 0;       // A: K offset per lane group
  int kbh  = (lane >= 16) ? 8 : 0;       // B: K offset per lane group
  int row  = tile * 16 + (lane & 15);    // memory row feeding column n
  v8f acc = {};
#pragma unroll
  for (int c = 0; c < 2; ++c) {          // K = 64 as 2 chunks of 32
    // A layout: element e -> K = (e&7) + (e&8 ? 16 : 0) (+ khl + c*32)
    const float* ap = keysp + mrow * WD + c * 32 + khl;
    v4f a0 = *(const v4f*)(ap);
    v4f a1 = *(const v4f*)(ap + 4);
    v4f a2 = *(const v4f*)(ap + 16);
    v4f a3 = *(const v4f*)(ap + 20);
    v16h a;
    a[0]=(_Float16)a0.x;  a[1]=(_Float16)a0.y;  a[2]=(_Float16)a0.z;  a[3]=(_Float16)a0.w;
    a[4]=(_Float16)a1.x;  a[5]=(_Float16)a1.y;  a[6]=(_Float16)a1.z;  a[7]=(_Float16)a1.w;
    a[8]=(_Float16)a2.x;  a[9]=(_Float16)a2.y;  a[10]=(_Float16)a2.z; a[11]=(_Float16)a2.w;
    a[12]=(_Float16)a3.x; a[13]=(_Float16)a3.y; a[14]=(_Float16)a3.z; a[15]=(_Float16)a3.w;
    // B layout: element e -> K = kbh + (e&7) + (e&8 ? 16 : 0), N = lane&15
    const float* bp = mem + (size_t)row * WD + c * 32 + kbh;
    v4f b0 = *(const v4f*)(bp);
    v4f b1 = *(const v4f*)(bp + 4);
    v4f b2 = *(const v4f*)(bp + 16);
    v4f b3 = *(const v4f*)(bp + 20);
    v16h b;
    b[0]=(_Float16)b0.x;  b[1]=(_Float16)b0.y;  b[2]=(_Float16)b0.z;  b[3]=(_Float16)b0.w;
    b[4]=(_Float16)b1.x;  b[5]=(_Float16)b1.y;  b[6]=(_Float16)b1.z;  b[7]=(_Float16)b1.w;
    b[8]=(_Float16)b2.x;  b[9]=(_Float16)b2.y;  b[10]=(_Float16)b2.z; b[11]=(_Float16)b2.w;
    b[12]=(_Float16)b3.x; b[13]=(_Float16)b3.y; b[14]=(_Float16)b3.z; b[15]=(_Float16)b3.w;
    acc = __builtin_amdgcn_wmma_f32_16x16x32_f16(false, a, false, b, (short)0, acc,
                                                 false, false);
  }
  if (lane < 16) {                       // D: lanes 0-15 hold M=0..7 in elements 0..7
    int ncol = tile * 16 + lane;
    float mn = mnorm[ncol];
#pragma unroll
    for (int w = 0; w < 8; ++w) {
      if (w < nkeys) {
        float st = strength ? strength[w] : 1.f;
        out[(size_t)w * MSLOT + ncol] = acc[w] / (knorm[w] * mn) * st;
      }
    }
  }
}

// ---- row softmax over MSLOT; mode=1 scales by read_mode content column ----
__global__ __launch_bounds__(1024)
void softmax_rows_kernel(const float* __restrict__ logits, float* __restrict__ out,
                         const float* __restrict__ ws, int mode) {
  __shared__ float red[1024];
  int hrow = blockIdx.x, t = threadIdx.x;
  const float* in = logits + (size_t)hrow * MSLOT;
  float mx = -1e30f;
  for (int m = t; m < MSLOT; m += 1024) mx = fmaxf(mx, in[m]);
  red[t] = mx; __syncthreads();
  for (int o = 512; o > 0; o >>= 1) { if (t < o) red[t] = fmaxf(red[t], red[t + o]); __syncthreads(); }
  float gmax = red[0]; __syncthreads();
  float s = 0.f;
  for (int m = t; m < MSLOT; m += 1024) s += __expf(in[m] - gmax);
  red[t] = s; __syncthreads();
  for (int o = 512; o > 0; o >>= 1) { if (t < o) red[t] += red[t + o]; __syncthreads(); }
  float inv = 1.f / red[0];
  float scale = mode ? ws[WS_RM + hrow * NRM + 2 * NWH] : 1.f;
  for (int m = t; m < MSLOT; m += 1024)
    out[(size_t)hrow * MSLOT + m] = scale * __expf(in[m] - gmax) * inv;
}

// ---- allocation weights: bitonic sort + multiply-scan in LDS, sequential over NW heads ----
__global__ __launch_bounds__(1024)
void alloc_kernel(float* __restrict__ ws, float* __restrict__ out) {
  __shared__ float s_val[MSLOT];
  __shared__ float s_cp[MSLOT];
  __shared__ float s_u[MSLOT];
  __shared__ unsigned short s_idx[MSLOT];
  int t = threadIdx.x;
  for (int q = 0; q < 4; ++q) s_u[t + q * 1024] = ws[WS_U + t + q * 1024];
  __syncthreads();
  for (int i = 0; i < NWH; ++i) {
    float wgate = ws[WS_AG + i] * ws[WS_WG + i];
    for (int q = 0; q < 4; ++q) {
      int m = t + q * 1024;
      s_val[m] = EPSX + (1.f - EPSX) * s_u[m];
      s_idx[m] = (unsigned short)m;
    }
    __syncthreads();
    // bitonic sort ascending (values + indices)
    for (int k = 2; k <= MSLOT; k <<= 1) {
      for (int j = k >> 1; j > 0; j >>= 1) {
        for (int q = 0; q < 4; ++q) {
          int a = t + q * 1024;
          int b = a ^ j;
          if (b > a) {
            bool up = ((a & k) == 0);
            float va = s_val[a], vb = s_val[b];
            if ((va > vb) == up) {
              s_val[a] = vb; s_val[b] = va;
              unsigned short ia = s_idx[a]; s_idx[a] = s_idx[b]; s_idx[b] = ia;
            }
          }
        }
        __syncthreads();
      }
    }
    // inclusive cumprod (Hillis-Steele)
    for (int q = 0; q < 4; ++q) { int m = t + q * 1024; s_cp[m] = s_val[m]; }
    __syncthreads();
    for (int off = 1; off < MSLOT; off <<= 1) {
      float nv[4];
      for (int q = 0; q < 4; ++q) {
        int m = t + q * 1024;
        nv[q] = (m >= off) ? s_cp[m - off] * s_cp[m] : s_cp[m];
      }
      __syncthreads();
      for (int q = 0; q < 4; ++q) s_cp[t + q * 1024] = nv[q];
      __syncthreads();
    }
    // sa = (1-su)*exclusive_cumprod, scatter to original slots (reuse s_cp as dense a)
    float sa[4]; int tgt[4];
    for (int q = 0; q < 4; ++q) {
      int m = t + q * 1024;
      float excl = (m == 0) ? 1.f : s_cp[m - 1];
      sa[q] = (1.f - s_val[m]) * excl;
      tgt[q] = s_idx[m];
    }
    __syncthreads();
    for (int q = 0; q < 4; ++q) s_cp[tgt[q]] = sa[q];
    __syncthreads();
    for (int q = 0; q < 4; ++q) {
      int m = t + q * 1024;
      float a = s_cp[m];
      ws[WS_ALLOC + i * MSLOT + m] = a;
      float u = s_u[m];
      s_u[m] = u + (1.f - u) * wgate * a;
    }
    __syncthreads();
  }
  for (int q = 0; q < 4; ++q) out[O_U + t + q * 1024] = s_u[t + q * 1024];
}

// ---- ww_new + prec_new (needs row sum of ww_new) ----
__global__ __launch_bounds__(1024)
void ww_prec_kernel(const float* __restrict__ ws, const float* __restrict__ prec_old,
                    float* __restrict__ out) {
  __shared__ float red[1024];
  int w = blockIdx.x, t = threadIdx.x;
  float ag = ws[WS_AG + w], wg = ws[WS_WG + w];
  float vals[4]; float local = 0.f;
  for (int q = 0; q < 4; ++q) {
    int m = t + q * 1024;
    float a = ws[WS_ALLOC + w * MSLOT + m];
    float c = ws[WS_WCONT + w * MSLOT + m];
    float v = wg * (ag * a + (1.f - ag) * c);
    vals[q] = v;
    out[O_WW + w * MSLOT + m] = v;
    local += v;
  }
  red[t] = local; __syncthreads();
  for (int o = 512; o > 0; o >>= 1) { if (t < o) red[t] += red[t + o]; __syncthreads(); }
  float wsum = red[0];
  for (int q = 0; q < 4; ++q) {
    int m = t + q * 1024;
    out[O_PREC + w * MSLOT + m] = (1.f - wsum) * prec_old[w * MSLOT + m] + vals[q];
  }
}

// ---- mem_new = memory * prod_w(1 - ww*e) + sum_w ww*wv; also new row norms ----
__global__ void memnew_kernel(const float* __restrict__ memory, float* __restrict__ ws,
                              float* __restrict__ out) {
  __shared__ float red[WD];
  int m = blockIdx.x, d = threadIdx.x;
  float w0 = out[O_WW + m];
  float w1 = out[O_WW + MSLOT + m];
  float e0 = ws[WS_EV + d],      e1 = ws[WS_EV + WD + d];
  float v0 = ws[WS_WV + d],      v1 = ws[WS_WV + WD + d];
  float val = memory[(size_t)m * WD + d] * (1.f - w0 * e0) * (1.f - w1 * e1)
            + w0 * v0 + w1 * v1;
  out[O_MEM + (size_t)m * WD + d] = val;
  red[d] = val * val; __syncthreads();
  for (int o = 32; o > 0; o >>= 1) { if (d < o) red[d] += red[d + o]; __syncthreads(); }
  if (d == 0) ws[WS_MN1 + m] = sqrtf(red[0]);
}

// ---- read_words[r][d] = sum_m rw_new[r][m] * mem_new[m][d] ----
__global__ void readwords_kernel(const float* __restrict__ rw_new,
                                 const float* __restrict__ mem_new,
                                 float* __restrict__ out_readw) {
  __shared__ float red[256];
  int r = blockIdx.x >> 6;
  int d = blockIdx.x & 63;
  int t = threadIdx.x;
  float s = 0.f;
  for (int m = t; m < MSLOT; m += 256)
    s += rw_new[(size_t)r * MSLOT + m] * mem_new[(size_t)m * WD + d];
  red[t] = s; __syncthreads();
  for (int o = 128; o > 0; o >>= 1) { if (t < o) red[t] += red[t + o]; __syncthreads(); }
  if (t == 0) out_readw[r * WD + d] = red[0];
}

// ---- dominant bandwidth kernel: link_new = (1 - wi - wj)*link + wi*pj, diag=0 ----
// link stream is ~268MB: use non-temporal load/store so the reused 16KB
// ww_new/precedence rows stay resident in cache while link streams through.
__global__ __launch_bounds__(1024)
void link_kernel(const float* __restrict__ link, const float* __restrict__ prec_old,
                 const float* __restrict__ ww_new, float* __restrict__ link_out) {
  int blk = blockIdx.x;            // (w, i) row
  int w = blk >> 12;
  int i = blk & (MSLOT - 1);
  float wi = ww_new[w * MSLOT + i];
  int j = threadIdx.x << 2;
  size_t rowbase = (size_t)blk * MSLOT;
  v4f l  = __builtin_nontemporal_load((const v4f*)(link + rowbase + j));
  v4f wj = *(const v4f*)(ww_new   + w * MSLOT + j);
  v4f pj = *(const v4f*)(prec_old + w * MSLOT + j);
  v4f o;
  o.x = (j + 0 == i) ? 0.f : (1.f - wi - wj.x) * l.x + wi * pj.x;
  o.y = (j + 1 == i) ? 0.f : (1.f - wi - wj.y) * l.y + wi * pj.y;
  o.z = (j + 2 == i) ? 0.f : (1.f - wi - wj.z) * l.z + wi * pj.z;
  o.w = (j + 3 == i) ? 0.f : (1.f - wi - wj.w) * l.w + wi * pj.w;
  __builtin_nontemporal_store(o, (v4f*)(link_out + rowbase + j));
}

extern "C" void kernel_launch(void* const* d_in, const int* in_sizes, int n_in,
                              void* d_out, int out_size, void* d_ws, size_t ws_size,
                              hipStream_t stream) {
  (void)in_sizes; (void)n_in; (void)out_size; (void)ws_size;
  const float* x      = (const float*)d_in[0];
  const float* memory = (const float*)d_in[1];
  const float* rw_old = (const float*)d_in[2];
  const float* ww_old = (const float*)d_in[3];
  const float* link   = (const float*)d_in[4];
  const float* prec   = (const float*)d_in[5];
  const float* usage  = (const float*)d_in[6];
  float* out = (float*)d_out;
  float* ws  = (float*)d_ws;

  struct P { int widx; int od; int wsofs; int act; };
  const P projs[9] = {
    { 7, NWH * WD, WS_WV,   0},   // write_vector
    { 9, NWH * WD, WS_EV,   1},   // erase (sigmoid)
    {11, RH,       WS_FG,   1},   // free_gate
    {13, NWH,      WS_AG,   1},   // allocation_gate
    {15, NWH,      WS_WG,   1},   // write_gate
    {17, RH * NRM, WS_RM,   1},   // read_mode
    {19, NWH * WD, WS_WK,   1},   // write_keys (sigmoid)
    {21, NWH,      WS_WSTR, 0},   // write_strengths
    {23, RH * WD,  WS_RK,   0},   // read_keys
  };
  for (int p = 0; p < 9; ++p)
    proj_kernel<<<projs[p].od, 256, 0, stream>>>(
        x, (const float*)d_in[projs[p].widx], (const float*)d_in[projs[p].widx + 1],
        ws + projs[p].wsofs, projs[p].od, projs[p].act);

  padkeys_kernel<<<1, 1024, 0, stream>>>(ws);
  usage_kernel<<<MSLOT / 256, 256, 0, stream>>>(usage, ww_old, rw_old, ws, ws + WS_U);
  rownorm_kernel<<<MSLOT, WD, 0, stream>>>(memory, ws + WS_MN0);

  cosine_wmma_kernel<<<MSLOT / 16, 32, 0, stream>>>(ws + WS_WKP, ws + WS_KNW, NWH,
                                                    memory, ws + WS_MN0, ws + WS_WSTR,
                                                    ws + WS_WLOGIT);
  softmax_rows_kernel<<<NWH, 1024, 0, stream>>>(ws + WS_WLOGIT, ws + WS_WCONT, ws, 0);

  alloc_kernel<<<1, 1024, 0, stream>>>(ws, out);
  ww_prec_kernel<<<NWH, 1024, 0, stream>>>(ws, prec, out);
  memnew_kernel<<<MSLOT, WD, 0, stream>>>(memory, ws, out);

  cosine_wmma_kernel<<<MSLOT / 16, 32, 0, stream>>>(ws + WS_RKP, ws + WS_KNR, RH,
                                                    out + O_MEM, ws + WS_MN1, nullptr,
                                                    ws + WS_RLOGIT);
  softmax_rows_kernel<<<RH, 1024, 0, stream>>>(ws + WS_RLOGIT, out + O_RW, ws, 1);
  readwords_kernel<<<RH * WD, 256, 0, stream>>>(out + O_RW, out + O_MEM, out + O_READW);

  link_kernel<<<NWH * MSLOT, 1024, 0, stream>>>(link, prec, out + O_WW, out + O_LINK);
}
